// LoraMoEBlock_7181185319314
// MI455X (gfx1250) — matrix operand order repaired and checked by reading, not compiled, not run on hardware
//
#include <hip/hip_runtime.h>
#include <hip/hip_bf16.h>
#include <math.h>

typedef _Float16 half_t;
typedef __attribute__((ext_vector_type(16))) _Float16 v16h;
typedef __attribute__((ext_vector_type(8)))  _Float16 v8h;
typedef __attribute__((ext_vector_type(8)))  float    v8f;

#define CCH 128
#define C4  32
#define HH  120
#define WW  160
#define NB  8
#define HWP (HH*WW)      /* 19200 */
#define MT  208          /* padded fused 3x3 output channels (13 tiles of 16) */
#define MTILES 13
#define KC3 36           /* 1152 / 32 */
#define EPSV 1e-5f
#define NEGINF -1e10f

struct Params {
  const float* x;
  const float *conv1_w, *conv1_b;
  const float *sa1_w, *sa1_b, *sa_ln_g, *sa_ln_b;
  const float *sasp_w, *sasp_b, *sa_lnsp_g, *sa_lnsp_b;
  const float *sa2_w, *sa2_b, *sa3_w, *sa3_b;
  const float *router_w, *router_b;
  const float *loraW[4], *loraWb[4], *loraM[4], *loraMb[4];
  const float *speW, *speWb, *speM, *speMb;
  const float *shareW, *shareB;
  const float *bnall_g, *bnall_b, *bnall_m, *bnall_v;
  const float *bn_g, *bn_b, *bn_m, *bn_v;
  const float *conv3_w, *conv3_b;
};

struct Work {
  half_t* out_ori;   // [NB][HWP][CCH]  f16 (NHWC)
  half_t* t;         // [NB][HWP][MT]   f16 (pixel-major)
  half_t* Acat;      // fused 3x3 weight frags: [13][36][32][16]
  half_t* A1;        // conv1 frags: [8][4][32][16]
  half_t* A3;        // conv3 frags: [8][4][32][16]
  half_t* Wc;        // per-sample mix frags: [NB][8][3][32][16]
  float*  bcat;      // [MT] 3x3 biases (lora/spe only)
  float*  btot;      // [NB][CCH] fused bias
  float*  inv2;      // [CCH] bn scale
  float*  osum;      // [NB][CCH] spatial sum of out_ori
  float*  fsum;      // [NB][C4]  spatial sum of sa3 features
};

__device__ __forceinline__ v8f wmma16(v16h a, v16h b, v8f c) {
  return __builtin_amdgcn_wmma_f32_16x16x32_f16(false, a, false, b, (short)0, c, false, false);
}
__device__ __forceinline__ float gelu_exact(float x) {
  return 0.5f * x * (1.0f + erff(x * 0.70710678118654752f));
}
// A-fragment K offset for element j, lane L (contiguous: k = 16*(L/16) + j, matches B loads)
__device__ __forceinline__ int afrag_k(int j, int lane) {
  return ((lane < 16) ? 0 : 16) + j;
}

// ---------------- weight packing + accumulator zeroing ----------------
__global__ void k_pack(Params p, Work w) {
  int tid = blockIdx.x * blockDim.x + threadIdx.x;
  int stride = gridDim.x * blockDim.x;
  for (int i = tid; i < NB*CCH; i += stride) w.osum[i] = 0.f;
  for (int i = tid; i < NB*C4;  i += stride) w.fsum[i] = 0.f;
  for (int m = tid; m < CCH; m += stride)
    w.inv2[m] = p.bn_g[m] * rsqrtf(p.bn_v[m] + EPSV);
  for (int m = tid; m < MT; m += stride) {
    float b = 0.f;
    if      (m < 4)  b = p.loraWb[0][m];
    else if (m < 12) b = p.loraWb[1][m-4];
    else if (m < 28) b = p.loraWb[2][m-12];
    else if (m < 60) b = p.loraWb[3][m-28];
    else if (m < 76) b = p.speWb[m-60];
    w.bcat[m] = b;                       // share bias folded into btot
  }
  // conv1 / conv3 A-fragments (8 mtiles x 4 kchunks x 32 lanes)
  for (int s = tid; s < 2048; s += stride) {
    const float* src = (s < 1024) ? p.conv1_w : p.conv3_w;
    half_t* dst = (s < 1024) ? w.A1 : w.A3;
    int s2 = s & 1023;
    int mt = s2 >> 7, kc = (s2 >> 5) & 3, lane = s2 & 31;
    int m = mt*16 + (lane & 15);
    alignas(16) half_t fr[16];
    #pragma unroll
    for (int j = 0; j < 16; ++j) {
      int k = kc*32 + afrag_k(j, lane);
      fr[j] = (half_t)src[m*CCH + k];
    }
    *(v8h*)(dst + s2*16)     = *(v8h*)&fr[0];
    *(v8h*)(dst + s2*16 + 8) = *(v8h*)&fr[8];
  }
  // fused 3x3 A-fragments: Wcat[m][k], k = tap*128 + c
  for (int s = tid; s < MTILES*KC3*32; s += stride) {
    int mt = s / (KC3*32), kc = (s / 32) % KC3, lane = s & 31;
    int m = mt*16 + (lane & 15);
    alignas(16) half_t fr[16];
    for (int j = 0; j < 16; ++j) {
      int k = kc*32 + afrag_k(j, lane);
      int tap = k >> 7, c = k & 127;
      float v = 0.f;
      if      (m < 4)   v = p.loraW[0][(m   )*CCH*9 + c*9 + tap];
      else if (m < 12)  v = p.loraW[1][(m-4 )*CCH*9 + c*9 + tap];
      else if (m < 28)  v = p.loraW[2][(m-12)*CCH*9 + c*9 + tap];
      else if (m < 60)  v = p.loraW[3][(m-28)*CCH*9 + c*9 + tap];
      else if (m < 76)  v = p.speW   [(m-60)*CCH*9 + c*9 + tap];
      else if (m < 204) v = p.shareW [(m-76)*CCH*9 + c*9 + tap];
      fr[j] = (half_t)v;
    }
    *(v8h*)(w.Acat + s*16)     = *(v8h*)&fr[0];
    *(v8h*)(w.Acat + s*16 + 8) = *(v8h*)&fr[8];
  }
}

// ---------------- conv1 1x1 (WMMA), x f32 NCHW -> out_ori f16 NHWC ----------------
__global__ void k_conv1(Params p, Work w) {
  __shared__ half_t lx[16*CCH];
  int bid = blockIdx.x;
  int n  = bid / (HWP/16);
  int p0 = (bid % (HWP/16)) * 16;
  int t = threadIdx.x, lane = t & 31, wv = t >> 5;
  #pragma unroll
  for (int it = 0; it < 8; ++it) {
    int c = (t >> 4) + it*16, i = t & 15;
    lx[i*CCH + c] = (half_t)p.x[((size_t)n*CCH + c)*HWP + p0 + i];
  }
  __syncthreads();
  v8f acc = {};
  #pragma unroll
  for (int kc = 0; kc < 4; ++kc) {
    v16h a = *(const v16h*)(w.A1 + (((wv*4 + kc)*32 + lane) << 4));
    v16h b = *(const v16h*)(lx + (lane & 15)*CCH + kc*32 + ((lane < 16) ? 0 : 16));
    acc = wmma16(a, b, acc);
  }
  int sel = (lane < 16) ? 0 : 8;
  int pg = p0 + (lane & 15);
  alignas(16) half_t o8[8];
  #pragma unroll
  for (int r = 0; r < 8; ++r) o8[r] = (half_t)(acc[r] + p.conv1_b[wv*16 + sel + r]);
  *(v8h*)(w.out_ori + ((size_t)(n*HWP + pg))*CCH + wv*16 + sel) = *(v8h*)o8;
}

// ---------------- per-pixel spatial attention + reductions ----------------
__global__ void k_sa(Params p, Work w) {
  __shared__ float lv[8*CCH];
  __shared__ float lf[8*C4];
  __shared__ float facc[C4];
  __shared__ float oacc[CCH];
  int bid = blockIdx.x;
  int n  = bid / (HWP/8);
  int p0 = (bid % (HWP/8)) * 8;
  int t = threadIdx.x, lane = t & 31, wv = t >> 5;
  if (t < C4)  facc[t] = 0.f;
  if (t < CCH) oacc[t] = 0.f;
  __syncthreads();
  int pg = p0 + wv;
  float v4[4];
  #pragma unroll
  for (int u = 0; u < 4; ++u) {
    float v = (float)w.out_ori[((size_t)(n*HWP + pg))*CCH + lane*4 + u];
    v4[u] = v;
    lv[wv*CCH + lane*4 + u] = v;
  }
  __syncthreads();
  float s1 = p.sa1_b[lane];
  for (int k = 0; k < CCH; ++k) s1 += p.sa1_w[lane*CCH + k] * lv[wv*CCH + k];
  float sum = s1;
  for (int m = 16; m >= 1; m >>= 1) sum += __shfl_xor(sum, m, 32);
  float mu = sum * (1.f/32.f);
  float d = s1 - mu;
  float vq = d*d;
  for (int m = 16; m >= 1; m >>= 1) vq += __shfl_xor(vq, m, 32);
  float var = vq * (1.f/32.f);
  float yn = d * rsqrtf(var + EPSV) * p.sa_ln_g[lane] + p.sa_ln_b[lane];
  lf[wv*C4 + lane] = gelu_exact(yn);
  __syncthreads();
  float s3 = p.sa3_b[lane];
  for (int k = 0; k < C4; ++k) s3 += p.sa3_w[lane*C4 + k] * lf[wv*C4 + k];
  atomicAdd(&facc[lane], s3);
  #pragma unroll
  for (int u = 0; u < 4; ++u) atomicAdd(&oacc[lane*4 + u], v4[u]);
  __syncthreads();
  if (t < C4)  atomicAdd(&w.fsum[n*C4  + t], facc[t]);
  if (t < CCH) atomicAdd(&w.osum[n*CCH + t], oacc[t]);
}

// ---------------- router: routing weights, BN folding, per-sample Wc frags ----------------
__global__ void k_router(Params p, Work w) {
  __shared__ float spv[CCH], tmp1[CCH], red[CCH];
  __shared__ float sp2[C4], featm[C4];
  __shared__ float prob[9], route[4];
  __shared__ float pmix;
  __shared__ float sA[CCH], sP[CCH];
  int n = blockIdx.x;
  int t = threadIdx.x;  // 128 threads
  spv[t] = w.osum[n*CCH + t] * (1.f/(float)HWP);
  __syncthreads();
  float acc = p.sasp_b[t];
  for (int k = 0; k < CCH; ++k) acc += p.sasp_w[t*CCH + k] * spv[k];
  red[t] = acc; __syncthreads();
  for (int s = 64; s >= 1; s >>= 1) { if (t < s) red[t] += red[t+s]; __syncthreads(); }
  float mu = red[0] * (1.f/CCH);
  __syncthreads();
  float d = acc - mu;
  red[t] = d*d; __syncthreads();
  for (int s = 64; s >= 1; s >>= 1) { if (t < s) red[t] += red[t+s]; __syncthreads(); }
  float var = red[0] * (1.f/CCH);
  __syncthreads();
  float yn = d * rsqrtf(var + EPSV) * p.sa_lnsp_g[t] + p.sa_lnsp_b[t];
  tmp1[t] = gelu_exact(yn);
  __syncthreads();
  if (t < C4) {
    float a2 = p.sa2_b[t];
    for (int k = 0; k < CCH; ++k) a2 += p.sa2_w[t*CCH + k] * tmp1[k];
    sp2[t] = a2;
    featm[t] = w.fsum[n*C4 + t] * (1.f/(float)HWP);
  }
  __syncthreads();
  if (t < 9) {
    float a = p.router_b[t];
    for (int k = 0; k < 64; ++k) {
      float rf = (k < 32) ? featm[k] : sp2[k-32];
      a += p.router_w[t*64 + k] * rf;
    }
    prob[t] = a;
  }
  __syncthreads();
  if (t == 0) {
    float rr[4]; for (int i = 0; i < 4; ++i) rr[i] = prob[i];
    int d1 = 0; for (int i = 1; i < 4; ++i) if (rr[i] < rr[d1]) d1 = i;
    int d2 = -1;
    for (int i = 0; i < 4; ++i) { if (i == d1) continue; if (d2 < 0 || rr[i] < rr[d2]) d2 = i; }
    rr[d1] = NEGINF; rr[d2] = NEGINF;
    float mx = fmaxf(fmaxf(rr[0], rr[1]), fmaxf(rr[2], rr[3]));
    float e[4], es = 0.f;
    for (int i = 0; i < 4; ++i) { e[i] = expf(rr[i] - mx); es += e[i]; }
    for (int i = 0; i < 4; ++i) route[i] = e[i] / es;
    pmix = prob[8];
  }
  __syncthreads();
  {
    float inv2   = p.bn_g[t]    * rsqrtf(p.bn_v[t]    + EPSV);
    float beta2  = p.bn_b[t]    - p.bn_m[t]   * inv2;
    float invall = p.bnall_g[t] * rsqrtf(p.bnall_v[t] + EPSV);
    float betall = p.bnall_b[t] - p.bnall_m[t] * invall;
    float bm = 0.f;
    for (int i = 0; i < 4; ++i) bm += route[i] * p.loraMb[i][t];
    w.btot[n*CCH + t] = inv2*(invall*bm + betall + pmix*p.speMb[t] + p.shareB[t]) + beta2;
    sA[t] = inv2 * invall;
    sP[t] = inv2 * pmix;
  }
  __syncthreads();
  half_t* Wcn = w.Wc + (size_t)n * (8*3*32*16);
  for (int s = t; s < 8*3*32; s += CCH) {
    int mt = s / 96, kc = (s/32) % 3, lane = s & 31;
    int m = mt*16 + (lane & 15);
    alignas(16) half_t fr[16];
    for (int j = 0; j < 16; ++j) {
      int k = kc*32 + afrag_k(j, lane);
      float v = 0.f;
      if      (k < 4)  v = sA[m]*route[0]*p.loraM[0][m*4  + k];
      else if (k < 12) v = sA[m]*route[1]*p.loraM[1][m*8  + (k-4)];
      else if (k < 28) v = sA[m]*route[2]*p.loraM[2][m*16 + (k-12)];
      else if (k < 60) v = sA[m]*route[3]*p.loraM[3][m*32 + (k-28)];
      else if (k < 76) v = sP[m]*p.speM[m*16 + (k-60)];
      fr[j] = (half_t)v;
    }
    *(v8h*)(Wcn + s*16)     = *(v8h*)&fr[0];
    *(v8h*)(Wcn + s*16 + 8) = *(v8h*)&fr[8];
  }
}

// ---------------- fused 3x3 conv: 208 output channels, WMMA implicit GEMM ----------------
// Patch staging uses CDNA5 async memory->LDS DMA (GLOBAL_LOAD_ASYNC_TO_LDS_B128,
// tracked by ASYNCcnt) for in-bounds pixels; halo zeros are written directly to LDS.
__global__ void k_conv3(Params p, Work w) {
  __shared__ half_t patch[54*CCH];   // 3 rows x 18 cols x 128 ch (sole LDS object -> base offset 0)
  int x0 = blockIdx.x * 16;
  int h  = blockIdx.y;
  int n  = blockIdx.z;
  int t = threadIdx.x, lane = t & 31, wv = t >> 5;
  // 54 slots of 256B; one wave moves 2 slots per op (16 lanes x 16B each)
  for (int s2 = wv; s2 < 27; s2 += 8) {
    int s  = s2*2 + (lane >> 4);
    int li = lane & 15;
    int gy = h + s/18 - 1;
    int gx = x0 + (s%18) - 1;
    unsigned loff = (unsigned)((s*CCH + li*8) * (int)sizeof(half_t));
    if (gy >= 0 && gy < HH && gx >= 0 && gx < WW) {
      unsigned long long ga = (unsigned long long)(w.out_ori +
          ((size_t)(n*HWP + gy*WW + gx))*CCH + li*8);
      asm volatile("global_load_async_to_lds_b128 %0, %1, off"
                   :: "v"(loff), "v"(ga) : "memory");
    } else {
      v8h z = {};
      *(v8h*)(patch + s*CCH + li*8) = z;
    }
  }
  asm volatile("s_wait_asynccnt 0x0" ::: "memory");
  __syncthreads();
  int px = lane & 15;
  int selB = (lane < 16) ? 0 : 16;
  size_t trow = ((size_t)(n*HWP + h*WW + x0 + px)) * MT;
  for (int mt = wv; mt < MTILES; mt += 8) {
    v8f acc = {};
    #pragma unroll 6
    for (int kc = 0; kc < KC3; ++kc) {
      int tap = kc >> 2;
      int cb  = (kc & 3) * 32;
      int slot = (tap/3)*18 + px + (tap%3);
      v16h b = *(const v16h*)(patch + slot*CCH + cb + selB);
      v16h a = *(const v16h*)(w.Acat + (((mt*KC3 + kc)*32 + lane) << 4));
      acc = wmma16(a, b, acc);
    }
    int ms = mt*16 + ((lane < 16) ? 0 : 8);
    alignas(16) half_t o8[8];
    #pragma unroll
    for (int r = 0; r < 8; ++r) o8[r] = (half_t)(acc[r] + w.bcat[ms + r]);
    *(v8h*)(w.t + trow + ms) = *(v8h*)o8;
  }
}

// ---------------- combine: Wc GEMM + share + bias -> GELU -> conv3 GEMM ----------------
__global__ void k_comb(Params p, Work w, float* out) {
  __shared__ half_t gbuf[16*CCH];
  int bid = blockIdx.x;
  int n  = bid / (HWP/16);
  int p0 = (bid % (HWP/16)) * 16;
  int t = threadIdx.x, lane = t & 31, wv = t >> 5;
  int px = lane & 15;
  int selB = (lane < 16) ? 0 : 16;
  int selD = (lane < 16) ? 0 : 8;
  int pg = p0 + px;
  const half_t* trow = w.t + ((size_t)(n*HWP + pg)) * MT;
  const half_t* Wcn  = w.Wc + (size_t)n * (8*3*32*16);
  v8f acc = {};
  #pragma unroll
  for (int kc = 0; kc < 3; ++kc) {
    v16h a = *(const v16h*)(Wcn + (((wv*3 + kc)*32 + lane) << 4));
    v16h b = *(const v16h*)(trow + kc*32 + selB);
    acc = wmma16(a, b, acc);
  }
  alignas(16) half_t g8[8];
  #pragma unroll
  for (int r = 0; r < 8; ++r) {
    int m = wv*16 + selD + r;
    float tsh = (float)trow[76 + m];
    float u = acc[r] + w.inv2[m]*tsh + w.btot[n*CCH + m];
    g8[r] = (half_t)gelu_exact(u);
  }
  *(v8h*)(gbuf + px*CCH + wv*16 + selD) = *(v8h*)g8;
  __syncthreads();
  v8f acc2 = {};
  #pragma unroll
  for (int kc = 0; kc < 4; ++kc) {
    v16h a = *(const v16h*)(w.A3 + (((wv*4 + kc)*32 + lane) << 4));
    v16h b = *(const v16h*)(gbuf + px*CCH + kc*32 + selB);
    acc2 = wmma16(a, b, acc2);
  }
  #pragma unroll
  for (int r = 0; r < 8; ++r) {
    int m = wv*16 + selD + r;
    out[((size_t)n*CCH + m)*HWP + pg] = acc2[r] + p.conv3_b[m];
  }
}

extern "C" void kernel_launch(void* const* d_in, const int* in_sizes, int n_in,
                              void* d_out, int out_size, void* d_ws, size_t ws_size,
                              hipStream_t stream) {
  const float* const* in = (const float* const*)d_in;
  Params p;
  int i = 0;
  p.x = in[i++];
  p.conv1_w = in[i++]; p.conv1_b = in[i++];
  p.sa1_w = in[i++]; p.sa1_b = in[i++];
  p.sa_ln_g = in[i++]; p.sa_ln_b = in[i++];
  p.sasp_w = in[i++]; p.sasp_b = in[i++];
  p.sa_lnsp_g = in[i++]; p.sa_lnsp_b = in[i++];
  p.sa2_w = in[i++]; p.sa2_b = in[i++];
  p.sa3_w = in[i++]; p.sa3_b = in[i++];
  p.router_w = in[i++]; p.router_b = in[i++];
  for (int j = 0; j < 4; ++j) {
    p.loraW[j] = in[i++]; p.loraWb[j] = in[i++];
    p.loraM[j] = in[i++]; p.loraMb[j] = in[i++];
  }
  p.speW = in[i++]; p.speWb = in[i++]; p.speM = in[i++]; p.speMb = in[i++];
  p.shareW = in[i++]; p.shareB = in[i++];
  p.bnall_g = in[i++]; p.bnall_b = in[i++]; p.bnall_m = in[i++]; p.bnall_v = in[i++];
  p.bn_g = in[i++]; p.bn_b = in[i++]; p.bn_m = in[i++]; p.bn_v = in[i++];
  p.conv3_w = in[i++]; p.conv3_b = in[i++];

  char* base = (char*)d_ws;
  size_t off = 0;
  auto alloc = [&](size_t bytes) -> void* {
    void* r = base + off;
    off = (off + bytes + 255) & ~(size_t)255;
    return r;
  };
  Work w;
  w.out_ori = (half_t*)alloc((size_t)NB*HWP*CCH*sizeof(half_t));
  w.t       = (half_t*)alloc((size_t)NB*HWP*MT*sizeof(half_t));
  w.Acat    = (half_t*)alloc((size_t)MTILES*KC3*32*16*sizeof(half_t));
  w.A1      = (half_t*)alloc((size_t)8*4*32*16*sizeof(half_t));
  w.A3      = (half_t*)alloc((size_t)8*4*32*16*sizeof(half_t));
  w.Wc      = (half_t*)alloc((size_t)NB*8*3*32*16*sizeof(half_t));
  w.bcat    = (float*)alloc(MT*sizeof(float));
  w.btot    = (float*)alloc((size_t)NB*CCH*sizeof(float));
  w.inv2    = (float*)alloc(CCH*sizeof(float));
  w.osum    = (float*)alloc((size_t)NB*CCH*sizeof(float));
  w.fsum    = (float*)alloc((size_t)NB*C4*sizeof(float));
  (void)ws_size; (void)in_sizes; (void)n_in; (void)out_size;

  k_pack  <<<80, 256, 0, stream>>>(p, w);
  k_conv1 <<<NB*HWP/16, 256, 0, stream>>>(p, w);
  k_sa    <<<NB*HWP/8, 256, 0, stream>>>(p, w);
  k_router<<<NB, 128, 0, stream>>>(p, w);
  k_conv3 <<<dim3(WW/16, HH, NB), 256, 0, stream>>>(p, w);
  k_comb  <<<NB*HWP/16, 256, 0, stream>>>(p, w, (float*)d_out);
}
